// ScaledDotProductAttention_5626407157993
// MI455X (gfx1250) — matrix operand order repaired and checked by reading, not compiled, not run on hardware
//
#include <hip/hip_runtime.h>
#include <hip/hip_bf16.h>

typedef __attribute__((ext_vector_type(16))) _Float16 v16h;
typedef __attribute__((ext_vector_type(8)))  _Float16 v8h;
typedef __attribute__((ext_vector_type(4)))  _Float16 v4h;
typedef __attribute__((ext_vector_type(8)))  float    v8f;

#define ATT_B   16
#define ATT_L   2048
#define ATT_D   128
// 1/sqrt(128) * log2(e): softmax carried in log2 domain (v_exp_f32 is exp2)
#define SCALE_L2 (0.08838834764831845f * 1.4426950408889634f)

#define KTILE 64
#define KT_S 136   // 128 + 8 halves pad  (row = 272B, 16B aligned)
#define VT_S 72    // 64 + 8 halves pad   (row = 144B, 16B aligned)
#define PW_S 72    // 64 + 8 halves pad

__device__ __forceinline__ v16h cat16(v8h lo, v8h hi) {
    v16h r;
#pragma unroll
    for (int i = 0; i < 8; ++i) { r[i] = lo[i]; r[i + 8] = hi[i]; }
    return r;
}

__device__ __forceinline__ v16h ld_lds16(const _Float16* p) {
    v8h lo = *(const v8h*)p;
    v8h hi = *(const v8h*)(p + 8);
    return cat16(lo, hi);
}

__device__ __forceinline__ float redmax16(float x) {
#pragma unroll
    for (int m = 1; m < 16; m <<= 1) x = fmaxf(x, __shfl_xor(x, m, 32));
    return x;
}

__device__ __forceinline__ v8h cvt8(float4 a, float4 b) {
    v8h r;
    r[0] = (_Float16)a.x; r[1] = (_Float16)a.y; r[2] = (_Float16)a.z; r[3] = (_Float16)a.w;
    r[4] = (_Float16)b.x; r[5] = (_Float16)b.y; r[6] = (_Float16)b.z; r[7] = (_Float16)b.w;
    return r;
}

__global__ __launch_bounds__(128, 1)
void fa_fwd_kernel(const float* __restrict__ q, const float* __restrict__ k,
                   const float* __restrict__ v, const int* __restrict__ mask,
                   float* __restrict__ out) {
    __shared__ _Float16 Kt[KTILE * KT_S];         // K tile, [key][d], f16
    __shared__ _Float16 Vt[ATT_D * VT_S];         // V tile transposed, [dv][key], f16
    __shared__ _Float16 Pw[4 * 2 * 16 * PW_S];    // per-wave, per-Mtile P staging, f16

    const int b     = blockIdx.y;
    const int lane  = threadIdx.x & 31;
    const int wv    = threadIdx.x >> 5;
    const int ln16  = lane & 15;
    const int hh    = lane >> 4;                  // which 16-lane half
    const int qrow0 = blockIdx.x * 128 + wv * 32; // 32 query rows per wave (2 M-tiles)

    // ---- preload Q: 2 M-tiles x 4 A-fragments (16x32 f16), f32->f16 on the fly
    v16h qa[2][4];
#pragma unroll
    for (int mi = 0; mi < 2; ++mi) {
        const float* qp = q + ((size_t)b * ATT_L + qrow0 + 16 * mi + ln16) * ATT_D;
#pragma unroll
        for (int c = 0; c < 4; ++c) {
            const int d0 = 32 * c + 16 * hh;       // run 1: K = d0..d0+7
            const int d1 = 32 * c + 16 * hh + 8;   // run 2: K = d1..d1+7
            float4 f0 = *(const float4*)(qp + d0);
            float4 f1 = *(const float4*)(qp + d0 + 4);
            float4 f2 = *(const float4*)(qp + d1);
            float4 f3 = *(const float4*)(qp + d1 + 4);
            qa[mi][c] = cat16(cvt8(f0, f1), cvt8(f2, f3));
        }
    }

    // constant all-ones B-fragment: P x ones = per-row sums (lands in C-layout)
    v16h onesB;
#pragma unroll
    for (int i = 0; i < 16; ++i) onesB[i] = (_Float16)1.0f;

    v8f acc[2][8];
    v8f vzero = {};
    float mrow[2][8], lrow[2][8];   // running max (log2 units) and sum
#pragma unroll
    for (int mi = 0; mi < 2; ++mi) {
#pragma unroll
        for (int d = 0; d < 8; ++d) acc[mi][d] = vzero;
#pragma unroll
        for (int r = 0; r < 8; ++r) { mrow[mi][r] = -INFINITY; lrow[mi][r] = 0.0f; }
    }

    _Float16* pw = &Pw[wv * 2 * 16 * PW_S];
    const int t_row = threadIdx.x >> 1;   // 0..63
    const int t_seg = threadIdx.x & 1;    // 0..1 (each = 64 floats)

    for (int kb = 0; kb < ATT_L; kb += KTILE) {
        __syncthreads();   // previous iteration's tile reads done

        // ---- cooperative stage: K tile (row-major f16) and V tile (transposed f16)
        {
            const float* gk = k + ((size_t)b * ATT_L + kb + t_row) * ATT_D + t_seg * 64;
            _Float16* dst = &Kt[t_row * KT_S + t_seg * 64];
#pragma unroll
            for (int i = 0; i < 16; ++i) {
                float4 f = ((const float4*)gk)[i];
                v4h h;
                h[0] = (_Float16)f.x; h[1] = (_Float16)f.y;
                h[2] = (_Float16)f.z; h[3] = (_Float16)f.w;
                *(v4h*)(dst + 4 * i) = h;
            }
            const float* gv = v + ((size_t)b * ATT_L + kb + t_row) * ATT_D + t_seg * 64;
#pragma unroll
            for (int i = 0; i < 16; ++i) {
                float4 f = ((const float4*)gv)[i];
                const int dv = t_seg * 64 + 4 * i;
                Vt[(dv + 0) * VT_S + t_row] = (_Float16)f.x;
                Vt[(dv + 1) * VT_S + t_row] = (_Float16)f.y;
                Vt[(dv + 2) * VT_S + t_row] = (_Float16)f.z;
                Vt[(dv + 3) * VT_S + t_row] = (_Float16)f.w;
            }
            if (kb + KTILE < ATT_L) {   // pull next tile toward L2/L0
                __builtin_prefetch(gk + KTILE * ATT_D, 0, 1);
                __builtin_prefetch(gv + KTILE * ATT_D, 0, 1);
            }
        }
        __syncthreads();   // tile ready

        // ---- S = Q * K^T : 8 independent chains, B-frags staged per d-chunk
        v8f s[2][4];
#pragma unroll
        for (int mi = 0; mi < 2; ++mi)
#pragma unroll
            for (int si = 0; si < 4; ++si) s[mi][si] = vzero;
#pragma unroll
        for (int c = 0; c < 4; ++c) {
            v16h kf[4];
#pragma unroll
            for (int si = 0; si < 4; ++si)
                kf[si] = ld_lds16(&Kt[(16 * si + ln16) * KT_S + 32 * c + 16 * hh]);
#pragma unroll
            for (int mi = 0; mi < 2; ++mi)
#pragma unroll
                for (int si = 0; si < 4; ++si)
                    s[mi][si] = __builtin_amdgcn_wmma_f32_16x16x32_f16(
                        false, qa[mi][c], false, kf[si], (short)0, s[mi][si], false, false);
        }

        float mk[4];
#pragma unroll
        for (int si = 0; si < 4; ++si)
            mk[si] = (float)mask[(size_t)b * ATT_L + kb + 16 * si + ln16];

        // ---- online softmax in log2 domain (row M = r + 8*hh; lane = one column)
        float p[2][4][8], scl[2][8];
#pragma unroll
        for (int mi = 0; mi < 2; ++mi) {
#pragma unroll
            for (int r = 0; r < 8; ++r) {
                float x0 = s[mi][0][r] * SCALE_L2;
                float x1 = s[mi][1][r] * SCALE_L2;
                float x2 = s[mi][2][r] * SCALE_L2;
                float x3 = s[mi][3][r] * SCALE_L2;
                float tm = redmax16(fmaxf(fmaxf(x0, x1), fmaxf(x2, x3)));
                float mn = fmaxf(mrow[mi][r], tm);
                float e0 = __builtin_amdgcn_exp2f(x0 - mn) * mk[0];   // mask post-exp
                float e1 = __builtin_amdgcn_exp2f(x1 - mn) * mk[1];
                float e2 = __builtin_amdgcn_exp2f(x2 - mn) * mk[2];
                float e3 = __builtin_amdgcn_exp2f(x3 - mn) * mk[3];
                float sc = __builtin_amdgcn_exp2f(mrow[mi][r] - mn);
                mrow[mi][r] = mn;
                scl[mi][r] = sc;
                p[mi][0][r] = e0; p[mi][1][r] = e1; p[mi][2][r] = e2; p[mi][3][r] = e3;
            }
#pragma unroll
            for (int d = 0; d < 8; ++d)
#pragma unroll
                for (int r = 0; r < 8; ++r) acc[mi][d][r] *= scl[mi][r];
        }

        // ---- C-layout -> A-layout for P (16x64) via per-wave LDS bounce
#pragma unroll
        for (int mi = 0; mi < 2; ++mi) {
            _Float16* pwm = pw + mi * 16 * PW_S;
#pragma unroll
            for (int r = 0; r < 8; ++r) {
                const int row = r + 8 * hh;
#pragma unroll
                for (int si = 0; si < 4; ++si)
                    pwm[row * PW_S + 16 * si + ln16] = (_Float16)p[mi][si][r];
            }
        }
        asm volatile("s_wait_dscnt 0" ::: "memory");
        v16h pa[2][2];   // [mi][Kp-chunk of 32 keys]
#pragma unroll
        for (int mi = 0; mi < 2; ++mi) {
            const _Float16* pwm = pw + mi * 16 * PW_S;
#pragma unroll
            for (int t = 0; t < 2; ++t) {
                v8h plo = *(const v8h*)(pwm + ln16 * PW_S + 32 * t + 16 * hh);
                v8h phi = *(const v8h*)(pwm + ln16 * PW_S + 32 * t + 16 * hh + 8);
                pa[mi][t] = cat16(plo, phi);
            }
        }

        // ---- row sums on the matrix pipe: ss = P x ones (masked P, f16-consistent)
        v8f ss[2];
#pragma unroll
        for (int mi = 0; mi < 2; ++mi) {
            ss[mi] = __builtin_amdgcn_wmma_f32_16x16x32_f16(
                false, pa[mi][0], false, onesB, (short)0, vzero, false, false);
            ss[mi] = __builtin_amdgcn_wmma_f32_16x16x32_f16(
                false, pa[mi][1], false, onesB, (short)0, ss[mi], false, false);
        }

        // ---- O += P * V : per d-chunk, 2 V-frags feed 4 WMMAs (16 chains total)
#pragma unroll
        for (int d = 0; d < 8; ++d) {
            v16h vf0 = ld_lds16(&Vt[(16 * d + ln16) * VT_S + 16 * hh]);
            v16h vf1 = ld_lds16(&Vt[(16 * d + ln16) * VT_S + 32 + 16 * hh]);
#pragma unroll
            for (int mi = 0; mi < 2; ++mi) {
                acc[mi][d] = __builtin_amdgcn_wmma_f32_16x16x32_f16(
                    false, pa[mi][0], false, vf0, (short)0, acc[mi][d], false, false);
                acc[mi][d] = __builtin_amdgcn_wmma_f32_16x16x32_f16(
                    false, pa[mi][1], false, vf1, (short)0, acc[mi][d], false, false);
            }
        }

        // ---- running-sum update from WMMA row sums
#pragma unroll
        for (int mi = 0; mi < 2; ++mi)
#pragma unroll
            for (int r = 0; r < 8; ++r)
                lrow[mi][r] = lrow[mi][r] * scl[mi][r] + ss[mi][r];
    }

    // ---- epilogue: divide by running sum, store f32
#pragma unroll
    for (int mi = 0; mi < 2; ++mi) {
        float inv[8];
#pragma unroll
        for (int r = 0; r < 8; ++r) inv[r] = 1.0f / lrow[mi][r];
#pragma unroll
        for (int d = 0; d < 8; ++d) {
#pragma unroll
            for (int r = 0; r < 8; ++r) {
                const size_t row = (size_t)b * ATT_L + qrow0 + 16 * mi + r + 8 * hh;
                out[row * ATT_D + 16 * d + ln16] = acc[mi][d][r] * inv[r];
            }
        }
    }
}

extern "C" void kernel_launch(void* const* d_in, const int* in_sizes, int n_in,
                              void* d_out, int out_size, void* d_ws, size_t ws_size,
                              hipStream_t stream) {
    (void)in_sizes; (void)n_in; (void)out_size; (void)d_ws; (void)ws_size;
    const float* q    = (const float*)d_in[0];
    const float* k    = (const float*)d_in[1];
    const float* v    = (const float*)d_in[2];
    const int*   mask = (const int*)d_in[3];
    float* out        = (float*)d_out;

    dim3 grid(ATT_L / 128, ATT_B);
    dim3 block(128);
    fa_fwd_kernel<<<grid, block, 0, stream>>>(q, k, v, mask, out);
}
